// MultiHeadedAttention_71528385348237
// MI455X (gfx1250) — matrix operand order repaired and checked by reading, not compiled, not run on hardware
//
#include <hip/hip_runtime.h>
#include <stdint.h>

typedef __attribute__((ext_vector_type(16))) _Float16 v16h;
typedef __attribute__((ext_vector_type(8)))  float    v8f;

constexpr int B  = 2;
constexpr int S  = 2048;
constexpr int D  = 1024;
constexpr int H  = 16;
constexpr int DK = 64;
constexpr int NROW = B * S;           // 4096

// ---------------- WMMA helpers (gfx1250 wave32, 16x16x32 f16 -> f32) ----------

__device__ __forceinline__ v8f wmma_f16(v16h a, v16h b, v8f c) {
  return __builtin_amdgcn_wmma_f32_16x16x32_f16(
      /*neg_a=*/false, a, /*neg_b=*/false, b,
      /*c_mod=*/(short)0, c, /*reuse_a=*/false, /*reuse_b=*/false);
}

union halfcvt { uint4 u[2]; v16h h; };

// A fragment: 16x32 f16 Per lane this code uses contiguous k (its own order, same as B):
// lanes 0-15 : k = 0..15 ; lanes 16-31 : k = 16..31.
// -> two ds_load_b128 per fragment.
__device__ __forceinline__ v16h load_a_frag(const _Float16* base, int stride, int lane) {
  int m  = lane & 15;
  int kh = (lane >> 4) & 1;
  const _Float16* row = base + m * stride + kh * 16;
  halfcvt c;
  c.u[0] = *(const uint4*)(row);        // k = kh*16 .. kh*16+7
  c.u[1] = *(const uint4*)(row + 8);    // k = kh*16+8 .. kh*16+15
  return c.h;
}

// B fragment (32x16) read from a contraction-minor LDS tile: base[n*stride + k].
// lanes 0-15 hold K=0..15, lanes 16-31 hold K=16..31; N = lane&15.
// Per lane: 16 contiguous halves -> two ds_load_b128.
__device__ __forceinline__ v16h load_bT_frag(const _Float16* base, int stride,
                                             int kbase, int lane) {
  int n  = lane & 15;
  int kh = (lane >> 4) & 1;
  const _Float16* p = base + n * stride + kbase + kh * 16;
  halfcvt c;
  c.u[0] = *(const uint4*)(p);
  c.u[1] = *(const uint4*)(p + 8);
  return c.h;
}

// ---- CDNA5 async global->LDS DMA (ASYNCcnt path, ISA 08_async_tensor §4) ----
// LDS byte address = low 32 bits of the generic shared pointer (aperture rule).

__device__ __forceinline__ void async_copy_b128(uint32_t lds_addr, const void* gaddr) {
  asm volatile("global_load_async_to_lds_b128 %0, %1, off"
               :: "v"(lds_addr), "v"(gaddr) : "memory");
}
__device__ __forceinline__ void wait_async0() {
  asm volatile("s_wait_asynccnt 0x0" ::: "memory");
}

// ---------------- Kernel 1: fused QKV projection ------------------------------
// out(f16, head-major [B,H,S,DK]) = f16( x @ W + bias ), x fp32 [NROW,D], W fp32 [D,D]
// blockIdx.z: 0->Q, 1->K, 2->V. Tile: 128x64 per block, 8 waves, BK=32.

__global__ __launch_bounds__(256) void qkv_proj_kernel(
    const float* __restrict__ qin, const float* __restrict__ kin,
    const float* __restrict__ vin,
    const float* __restrict__ wq, const float* __restrict__ bq,
    const float* __restrict__ wk, const float* __restrict__ bk,
    const float* __restrict__ wv, const float* __restrict__ bv,
    _Float16* __restrict__ qh, _Float16* __restrict__ kh,
    _Float16* __restrict__ vh) {
  constexpr int BM = 128, BN = 64, BK = 32;
  __shared__ _Float16 shX[BM * BK];     // [row][k]
  __shared__ _Float16 shWT[BN * BK];    // [n][k]  (contraction-minor)

  const int which = blockIdx.z;
  const float* x    = (which == 0) ? qin : (which == 1) ? kin : vin;
  const float* w    = (which == 0) ? wq  : (which == 1) ? wk  : wv;
  const float* bias = (which == 0) ? bq  : (which == 1) ? bk  : bv;
  _Float16*    dst  = (which == 0) ? qh  : (which == 1) ? kh  : vh;

  const int tid = threadIdx.x, wave = tid >> 5, lane = tid & 31;
  const int m0 = blockIdx.y * BM, n0 = blockIdx.x * BN;

  v8f acc[4] = {};

  for (int kt = 0; kt < D / BK; ++kt) {
    const int kbase = kt * BK;
    // stage X tile (128x32 fp32 -> f16), row-major: 1024 float4, 4 per thread
#pragma unroll
    for (int i = 0; i < 4; ++i) {
      int idx = tid + i * 256;
      int r = idx >> 3, c4 = idx & 7;
      float4 f = ((const float4*)(x + (size_t)(m0 + r) * D + kbase))[c4];
      _Float16* d = &shX[r * BK + c4 * 4];
      d[0] = (_Float16)f.x; d[1] = (_Float16)f.y;
      d[2] = (_Float16)f.z; d[3] = (_Float16)f.w;
    }
    // stage W tile (32x64 fp32 -> f16) TRANSPOSED into [n][k]: 512 float4, 2/thread
#pragma unroll
    for (int i = 0; i < 2; ++i) {
      int idx = tid + i * 256;
      int r = idx >> 4, c4 = idx & 15;       // r = k index, c4*4 = n offset
      float4 f = ((const float4*)(w + (size_t)(kbase + r) * D + n0))[c4];
      shWT[(c4 * 4 + 0) * BK + r] = (_Float16)f.x;
      shWT[(c4 * 4 + 1) * BK + r] = (_Float16)f.y;
      shWT[(c4 * 4 + 2) * BK + r] = (_Float16)f.z;
      shWT[(c4 * 4 + 3) * BK + r] = (_Float16)f.w;
    }
    // prefetch next W tile into GL2 (global_prefetch_b8)
    if (kt + 1 < D / BK)
      __builtin_prefetch(w + (size_t)(kbase + BK + (tid >> 3)) * D + n0 + (tid & 7) * 8, 0, 3);
    __syncthreads();

    v16h a = load_a_frag(&shX[(wave * 16) * BK], BK, lane);
#pragma unroll
    for (int nt = 0; nt < 4; ++nt) {
      v16h bf = load_bT_frag(&shWT[(nt * 16) * BK], BK, 0, lane);
      acc[nt] = wmma_f16(a, bf, acc[nt]);
    }
    __syncthreads();
  }

  // epilogue: add bias, scatter into [B,H,S,DK] f16
  const int halfid = lane >> 4, c16 = lane & 15;
#pragma unroll
  for (int nt = 0; nt < 4; ++nt) {
    int col = n0 + nt * 16 + c16;
    int h = col >> 6, d = col & 63;
    float bb = bias[col];
#pragma unroll
    for (int r = 0; r < 8; ++r) {
      int row = m0 + wave * 16 + r + halfid * 8;
      int bidx = row >> 11;          // / S
      int s    = row & (S - 1);
      dst[(((size_t)bidx * H + h) * S + s) * DK + d] = (_Float16)(acc[nt][r] + bb);
    }
  }
}

// ---------------- Kernel 2: flash attention ------------------------------------
// One block per (qtile of 64, head, batch). 4 waves x 16 query rows each.

__global__ __launch_bounds__(128) void attn_kernel(
    const _Float16* __restrict__ qh, const _Float16* __restrict__ kh,
    const _Float16* __restrict__ vh,
    const int* __restrict__ mask, const int* __restrict__ pad,
    _Float16* __restrict__ ctx) {
  constexpr float SCALE = 0.125f;   // 1/sqrt(64)
  __shared__ _Float16 shQ[64 * 64];        // [qrow][d]
  __shared__ _Float16 shK[64 * 64];        // [key][d]   (contraction-minor for QK^T)
  __shared__ _Float16 shVT[64 * 64];       // [d][key]   (contraction-minor for P.V)
  __shared__ _Float16 shP[4][16 * 64];     // per-wave [qrow][key]

  const int tid = threadIdx.x, wave = tid >> 5, lane = tid & 31;
  const int qt = blockIdx.x, h = blockIdx.y, b = blockIdx.z;
  const size_t headoff = ((size_t)b * H + h) * S * DK;
  const int q0 = qt * 64;

  const uint32_t qlds = (uint32_t)(uintptr_t)&shQ[0];
  const uint32_t klds = (uint32_t)(uintptr_t)&shK[0];

  // cooperative Q tile load via async DMA (8KB = 512 x b128, 4 per thread)
  {
    const char* qg = (const char*)(qh + headoff + (size_t)q0 * DK);
#pragma unroll
    for (int i = 0; i < 4; ++i) {
      int c = tid + i * 128;
      async_copy_b128(qlds + c * 16, qg + c * 16);
    }
    wait_async0();
  }
  __syncthreads();

  const int qr0 = wave * 16;
  const int halfid = lane >> 4, c16 = lane & 15;
  v16h aq0 = load_a_frag(&shQ[qr0 * 64], 64, lane);
  v16h aq1 = load_a_frag(&shQ[qr0 * 64 + 32], 64, lane);

  v8f o[4] = {};
  float mrow[8], lrow[8];
#pragma unroll
  for (int r = 0; r < 8; ++r) { mrow[r] = -1e30f; lrow[r] = 0.f; }

  for (int kt = 0; kt < S / 64; ++kt) {
    const int k0 = kt * 64;
    __syncthreads();   // previous iteration done reading shK/shVT
    {
      // K tile: raw copy via async DMA (4 x b128 per thread)
      const char* kg = (const char*)(kh + headoff + (size_t)k0 * DK);
#pragma unroll
      for (int i = 0; i < 4; ++i) {
        int c = tid + i * 128;
        async_copy_b128(klds + c * 16, kg + c * 16);
      }
      // V tile: transpose into [d][key] (2048 dwords, 16 per thread)
      const uint32_t* sv = (const uint32_t*)(vh + headoff + (size_t)k0 * DK);
#pragma unroll
      for (int i = 0; i < 16; ++i) {
        int idx = tid + i * 128;
        int key = idx >> 5, c = idx & 31;
        union { uint32_t u; _Float16 hh[2]; } cv;
        cv.u = sv[idx];
        shVT[(2 * c + 0) * 64 + key] = cv.hh[0];
        shVT[(2 * c + 1) * 64 + key] = cv.hh[1];
      }
      wait_async0();
    }
    __syncthreads();

    // scores: S = Q . K^T   (contraction over DK=64, two 32-chunks)
    v8f sc[4];
#pragma unroll
    for (int nt = 0; nt < 4; ++nt) {
      v8f s = {};
      s = wmma_f16(aq0, load_bT_frag(&shK[(nt * 16) * 64], 64, 0,  lane), s);
      s = wmma_f16(aq1, load_bT_frag(&shK[(nt * 16) * 64], 64, 32, lane), s);
      sc[nt] = s;
    }

    // scale + attention mask
#pragma unroll
    for (int nt = 0; nt < 4; ++nt) {
      int kcol = k0 + nt * 16 + c16;
#pragma unroll
      for (int r = 0; r < 8; ++r) {
        int qrow = q0 + qr0 + r + halfid * 8;
        float sval = sc[nt][r] * SCALE;
        int mv = mask[((size_t)b * S + qrow) * S + kcol];
        sc[nt][r] = (mv == 0) ? -1e30f : sval;
      }
    }

    // online softmax (row stats reduced across the 16-lane half via shfl_xor)
    float alpha[8];
#pragma unroll
    for (int r = 0; r < 8; ++r) {
      float rm = fmaxf(fmaxf(sc[0][r], sc[1][r]), fmaxf(sc[2][r], sc[3][r]));
#pragma unroll
      for (int off = 8; off >= 1; off >>= 1) rm = fmaxf(rm, __shfl_xor(rm, off, 32));
      float mnew = fmaxf(mrow[r], rm);
      alpha[r] = __expf(mrow[r] - mnew);
      float rs = 0.f;
#pragma unroll
      for (int nt = 0; nt < 4; ++nt) {
        float p = __expf(sc[nt][r] - mnew);
        sc[nt][r] = p;
        rs += p;
      }
#pragma unroll
      for (int off = 8; off >= 1; off >>= 1) rs += __shfl_xor(rs, off, 32);
      lrow[r] = lrow[r] * alpha[r] + rs;   // denominator includes padded keys
      mrow[r] = mnew;
    }

    // padding mask applied AFTER row-sum (reference zeroes post-softmax),
    // write P' f16 to wave-private LDS in A-major [row][key] layout
    _Float16* pw = shP[wave];
#pragma unroll
    for (int nt = 0; nt < 4; ++nt) {
      int kcol = k0 + nt * 16 + c16;
      float pm = (pad[b * S + kcol] == 0) ? 0.f : 1.f;
#pragma unroll
      for (int r = 0; r < 8; ++r)
        pw[(r + halfid * 8) * 64 + nt * 16 + c16] = (_Float16)(sc[nt][r] * pm);
    }
    asm volatile("s_wait_dscnt 0" ::: "memory");  // same-wave DS RAW before reload

    // rescale running output
#pragma unroll
    for (int nt = 0; nt < 4; ++nt)
#pragma unroll
      for (int r = 0; r < 8; ++r) o[nt][r] *= alpha[r];

    // O += P' . V  (contraction over 64 keys, two 32-chunks; V stored [d][key])
    v16h ap0 = load_a_frag(pw, 64, lane);
    v16h ap1 = load_a_frag(pw + 32, 64, lane);
#pragma unroll
    for (int nt = 0; nt < 4; ++nt) {
      o[nt] = wmma_f16(ap0, load_bT_frag(&shVT[(nt * 16) * 64], 64, 0,  lane), o[nt]);
      o[nt] = wmma_f16(ap1, load_bT_frag(&shVT[(nt * 16) * 64], 64, 32, lane), o[nt]);
    }
  }

  // normalize and write ctx f16 [B,S,D] with col = h*64 + d
#pragma unroll
  for (int r = 0; r < 8; ++r) {
    float inv = (lrow[r] > 0.f) ? 1.f / lrow[r] : 0.f;
    int qrow = q0 + qr0 + r + halfid * 8;
#pragma unroll
    for (int nt = 0; nt < 4; ++nt)
      ctx[((size_t)b * S + qrow) * D + h * DK + nt * 16 + c16] =
          (_Float16)(o[nt][r] * inv);
  }
}

// ---------------- Kernel 3: output projection ----------------------------------
// out(fp32 [NROW,D]) = ctx(f16) @ wo(fp32->f16) + bo

__global__ __launch_bounds__(256) void oproj_kernel(
    const _Float16* __restrict__ ctx, const float* __restrict__ wo,
    const float* __restrict__ bo, float* __restrict__ out) {
  constexpr int BM = 128, BN = 64, BK = 32;
  __shared__ _Float16 shA[BM * BK];     // [row][k]
  __shared__ _Float16 shWT[BN * BK];    // [n][k]

  const int tid = threadIdx.x, wave = tid >> 5, lane = tid & 31;
  const int m0 = blockIdx.y * BM, n0 = blockIdx.x * BN;

  v8f acc[4] = {};

  for (int kt = 0; kt < D / BK; ++kt) {
    const int kbase = kt * BK;
    // stage A tile (f16 copy, 2048 dwords / 256 threads)
#pragma unroll
    for (int i = 0; i < 8; ++i) {
      int idx = tid + i * 256;
      int r = idx >> 4, c = idx & 15;
      ((uint32_t*)shA)[r * 16 + c] =
          ((const uint32_t*)(ctx + (size_t)(m0 + r) * D + kbase))[c];
    }
    // stage W tile (fp32 -> f16) TRANSPOSED into [n][k]
#pragma unroll
    for (int i = 0; i < 2; ++i) {
      int idx = tid + i * 256;
      int r = idx >> 4, c4 = idx & 15;
      float4 f = ((const float4*)(wo + (size_t)(kbase + r) * D + n0))[c4];
      shWT[(c4 * 4 + 0) * BK + r] = (_Float16)f.x;
      shWT[(c4 * 4 + 1) * BK + r] = (_Float16)f.y;
      shWT[(c4 * 4 + 2) * BK + r] = (_Float16)f.z;
      shWT[(c4 * 4 + 3) * BK + r] = (_Float16)f.w;
    }
    if (kt + 1 < D / BK)
      __builtin_prefetch(wo + (size_t)(kbase + BK + (tid >> 3)) * D + n0 + (tid & 7) * 8, 0, 3);
    __syncthreads();

    v16h a = load_a_frag(&shA[(wave * 16) * BK], BK, lane);
#pragma unroll
    for (int nt = 0; nt < 4; ++nt) {
      v16h bf = load_bT_frag(&shWT[(nt * 16) * BK], BK, 0, lane);
      acc[nt] = wmma_f16(a, bf, acc[nt]);
    }
    __syncthreads();
  }

  const int halfid = lane >> 4, c16 = lane & 15;
#pragma unroll
  for (int nt = 0; nt < 4; ++nt) {
    int col = n0 + nt * 16 + c16;
    float bb = bo[col];
#pragma unroll
    for (int r = 0; r < 8; ++r) {
      int row = m0 + wave * 16 + r + halfid * 8;
      out[(size_t)row * D + col] = acc[nt][r] + bb;
    }
  }
}

// ---------------- launch --------------------------------------------------------

extern "C" void kernel_launch(void* const* d_in, const int* in_sizes, int n_in,
                              void* d_out, int out_size, void* d_ws, size_t ws_size,
                              hipStream_t stream) {
  const float* q    = (const float*)d_in[0];
  const float* k    = (const float*)d_in[1];
  const float* v    = (const float*)d_in[2];
  const int*   mask = (const int*)d_in[3];
  const int*   pad  = (const int*)d_in[4];
  const float* wq   = (const float*)d_in[5];
  const float* bq   = (const float*)d_in[6];
  const float* wk   = (const float*)d_in[7];
  const float* bk   = (const float*)d_in[8];
  const float* wv   = (const float*)d_in[9];
  const float* bv   = (const float*)d_in[10];
  const float* wo   = (const float*)d_in[11];
  const float* bo   = (const float*)d_in[12];

  const size_t NHALF = (size_t)B * H * S * DK;   // 4 Mi halves per buffer
  _Float16* qh  = (_Float16*)d_ws;
  _Float16* khp = qh  + NHALF;
  _Float16* vhp = khp + NHALF;
  _Float16* ctx = vhp + NHALF;

  dim3 g1(D / 64, NROW / 128, 3);
  qkv_proj_kernel<<<g1, 256, 0, stream>>>(q, k, v, wq, bq, wk, bk, wv, bv,
                                          qh, khp, vhp);

  dim3 g2(S / 64, H, B);
  attn_kernel<<<g2, 128, 0, stream>>>(qh, khp, vhp, mask, pad, ctx);

  dim3 g3(D / 64, NROW / 128);
  oproj_kernel<<<g3, 256, 0, stream>>>(ctx, wo, bo, (float*)d_out);
}
